// CausalSelfAttention_38328288150347
// MI455X (gfx1250) — matrix operand order repaired and checked by reading, not compiled, not run on hardware
//
#include <hip/hip_runtime.h>

typedef __bf16 bf16_t;
typedef __attribute__((ext_vector_type(16))) __bf16 v16bf;
typedef __attribute__((ext_vector_type(8)))  __bf16 v8bf;
typedef __attribute__((ext_vector_type(8)))  float  v8f;
typedef __attribute__((ext_vector_type(4)))  float  v4f;

#define TSEQ  2048
#define CDIM  1024
#define NHEAD 16
#define DHEAD 64
#define BATCH 2

static __device__ __forceinline__ v16bf cat16(v8bf lo, v8bf hi) {
  v16bf r;
#pragma unroll
  for (int i = 0; i < 8; ++i) { r[i] = lo[i]; r[i + 8] = hi[i]; }
  return r;
}

static __device__ __forceinline__ v8bf cvt8(v4f a, v4f b) {
  v8bf r;
#pragma unroll
  for (int i = 0; i < 4; ++i) { r[i] = (__bf16)a[i]; r[i + 4] = (__bf16)b[i]; }
  return r;
}

static __device__ __forceinline__ v8f v8f_zero() {
  v8f r;
#pragma unroll
  for (int i = 0; i < 8; ++i) r[i] = 0.0f;
  return r;
}

// --- CDNA5 async global->LDS copy (ASYNCcnt path), with safe fallback -------
#if __has_builtin(__builtin_amdgcn_global_load_async_to_lds_b128)
#define HAS_ASYNC_LDS 1
#else
#define HAS_ASYNC_LDS 0
#endif

typedef int i32x4 __attribute__((vector_size(16)));
typedef __attribute__((address_space(1))) i32x4 as1_i32x4;
typedef __attribute__((address_space(3))) i32x4 as3_i32x4;

static __device__ __forceinline__ void async_copy16(const bf16_t* gsrc, bf16_t* ldst) {
#if HAS_ASYNC_LDS
  __builtin_amdgcn_global_load_async_to_lds_b128(
      (as1_i32x4*)gsrc, (as3_i32x4*)ldst, 0, 0);
#else
  *(v8bf*)ldst = *(const v8bf*)gsrc;
#endif
}

static __device__ __forceinline__ void wait_async3() {
#if HAS_ASYNC_LDS
  asm volatile("s_wait_asynccnt 0x3" ::: "memory");
#endif
}
static __device__ __forceinline__ void wait_async0() {
#if HAS_ASYNC_LDS
  asm volatile("s_wait_asynccnt 0x0" ::: "memory");
#endif
}

// ---------------------------------------------------------------------------
// Kernel 0a: elementwise fp32 -> bf16 (for x activations)
// ---------------------------------------------------------------------------
__global__ __launch_bounds__(256) void cvt_kernel(const float* __restrict__ src,
                                                  bf16_t* __restrict__ dst, int n8)
{
  const int i = blockIdx.x * 256 + threadIdx.x;
  if (i < n8) {
    const float* s = src + (size_t)i * 8;
    *(v8bf*)(dst + (size_t)i * 8) = cvt8(*(const v4f*)s, *(const v4f*)(s + 4));
  }
}

// ---------------------------------------------------------------------------
// Kernel 0b: weight transpose+convert  w[k][n] fp32  ->  wt[n][k] bf16
// so GEMM B-fragments (column-contiguous along k) are straight b128 copies.
// ---------------------------------------------------------------------------
__global__ __launch_bounds__(256) void wtrans_kernel(
    const float* __restrict__ w, bf16_t* __restrict__ wt, int K, int N)
{
  __shared__ float tile[32][33];
  const int tx  = threadIdx.x & 31;
  const int ty4 = (threadIdx.x >> 5) * 4;
  const int k0 = blockIdx.y * 32;
  const int n0 = blockIdx.x * 32;
#pragma unroll
  for (int i = 0; i < 4; ++i)
    tile[ty4 + i][tx] = w[(size_t)(k0 + ty4 + i) * N + n0 + tx];
  __syncthreads();
#pragma unroll
  for (int i = 0; i < 4; ++i)
    wt[(size_t)(n0 + ty4 + i) * K + k0 + tx] = (__bf16)tile[tx][ty4 + i];
}

// ---------------------------------------------------------------------------
// Kernel 1: qkv = xb @ watT   (bf16 in via async-LDS double buffering)
//   q,k -> [B,H,T,D]   v -> [B,H,D,T] (transposed for PV B-fragments)
// Block: 256 thr (8 waves), tile 128(M) x 64(N), K-step 32, 2x LDS buffers.
// ---------------------------------------------------------------------------
__global__ __launch_bounds__(256) void qkv_gemm_kernel(
    const bf16_t* __restrict__ xb, const bf16_t* __restrict__ wat,
    bf16_t* __restrict__ qb, bf16_t* __restrict__ kb, bf16_t* __restrict__ vt)
{
  constexpr int K = CDIM;
  __shared__ __align__(16) bf16_t As[2][128][32];   // [m][k]
  __shared__ __align__(16) bf16_t Bs[2][64][32];    // [n][k]

  const int tid  = threadIdx.x;
  const int lane = tid & 31;
  const int wid  = tid >> 5;
  const int ll   = lane & 15;
  const int lh   = lane >> 4;
  const int waveM = wid & 3;
  const int waveN = wid >> 2;
  const int m0 = blockIdx.y * 128;
  const int n0 = blockIdx.x * 64;

  v8f acc[2][2];
#pragma unroll
  for (int i = 0; i < 2; ++i)
#pragma unroll
    for (int j = 0; j < 2; ++j) acc[i][j] = v8f_zero();

  auto stage = [&](int k0, int buf) {
#pragma unroll
    for (int p = 0; p < 2; ++p) {
      const int row = p * 64 + (tid >> 2);
      const int kc  = (tid & 3) * 8;
      async_copy16(xb + (size_t)(m0 + row) * K + k0 + kc, &As[buf][row][kc]);
    }
    const int n  = tid >> 2;
    const int kc = (tid & 3) * 8;
    async_copy16(wat + (size_t)(n0 + n) * K + k0 + kc, &Bs[buf][n][kc]);
  };

  stage(0, 0);
  for (int k0 = 0; k0 < K; k0 += 32) {
    const int cur = (k0 >> 5) & 1;
    if (k0 + 32 < K) { stage(k0 + 32, cur ^ 1); wait_async3(); }
    else             { wait_async0(); }
    __syncthreads();

    v16bf af[2], bfr[2];
#pragma unroll
    for (int mt = 0; mt < 2; ++mt) {
      const bf16_t* s = &As[cur][waveM * 32 + mt * 16 + ll][lh * 8];
      af[mt] = cat16(*(const v8bf*)s, *(const v8bf*)(s + 16));
    }
#pragma unroll
    for (int nt = 0; nt < 2; ++nt) {
      const bf16_t* s = &Bs[cur][waveN * 32 + nt * 16 + ll][lh * 8];
      bfr[nt] = cat16(*(const v8bf*)s, *(const v8bf*)(s + 16));
    }
#pragma unroll
    for (int mt = 0; mt < 2; ++mt)
#pragma unroll
      for (int nt = 0; nt < 2; ++nt)
        acc[mt][nt] = __builtin_amdgcn_wmma_f32_16x16x32_bf16(
            false, af[mt], false, bfr[nt], (short)0, acc[mt][nt], false, false);
    __syncthreads();
  }

  // epilogue: scatter bf16 into q/k [B,H,T,D] and v-transposed [B,H,D,T]
#pragma unroll
  for (int mt = 0; mt < 2; ++mt) {
#pragma unroll
    for (int nt = 0; nt < 2; ++nt) {
      const int n   = n0 + waveN * 32 + nt * 16 + ll;
      const int sec = n / CDIM;        // 0=q 1=k 2=v
      const int rem = n % CDIM;
      const int h   = rem / DHEAD;
      const int d   = rem % DHEAD;
#pragma unroll
      for (int r = 0; r < 8; ++r) {
        const int m = m0 + waveM * 32 + mt * 16 + r + 8 * lh;
        const int b = m >> 11;
        const int t = m & (TSEQ - 1);
        const size_t bhix = (size_t)(b * NHEAD + h);
        const bf16_t v = (__bf16)acc[mt][nt][r];
        if (sec == 0)      qb[(bhix * TSEQ + t) * DHEAD + d] = v;
        else if (sec == 1) kb[(bhix * TSEQ + t) * DHEAD + d] = v;
        else               vt[(bhix * DHEAD + d) * TSEQ + t] = v;
      }
    }
  }
}

// ---------------------------------------------------------------------------
// Kernel 2: flash attention.  Block = 128 thr (4 waves); wave owns 16 q rows.
// kv chunks of 32, online softmax, P redistributed via per-wave LDS tile.
// ---------------------------------------------------------------------------
__global__ __launch_bounds__(128) void flash_attn_kernel(
    const bf16_t* __restrict__ qb, const bf16_t* __restrict__ kbuf,
    const bf16_t* __restrict__ vt, bf16_t* __restrict__ yb)
{
  __shared__ __align__(16) bf16_t Ps[4][16][32];
  const int tid  = threadIdx.x;
  const int lane = tid & 31;
  const int wid  = tid >> 5;
  const int ll   = lane & 15;
  const int lh   = lane >> 4;
  const int bhix = blockIdx.y;              // b*16 + h
  const int b    = bhix >> 4;
  const int h    = bhix & (NHEAD - 1);
  const int q0   = blockIdx.x * 64 + wid * 16;

  const bf16_t* qp = qb   + (size_t)bhix * TSEQ * DHEAD;
  const bf16_t* kp = kbuf + (size_t)bhix * TSEQ * DHEAD;
  const bf16_t* vp = vt   + (size_t)bhix * DHEAD * TSEQ;

  // Q A-fragments for d-chunks [0,32) and [32,64)
  v16bf qf[2];
#pragma unroll
  for (int c = 0; c < 2; ++c) {
    const bf16_t* s = qp + (size_t)(q0 + ll) * DHEAD + c * 32 + lh * 8;
    qf[c] = cat16(*(const v8bf*)s, *(const v8bf*)(s + 16));
  }

  v8f o[4];
#pragma unroll
  for (int dt = 0; dt < 4; ++dt) o[dt] = v8f_zero();
  float mi[8], li[8];
#pragma unroll
  for (int r = 0; r < 8; ++r) { mi[r] = -1e30f; li[r] = 0.0f; }

  const float scale = 0.125f;   // 1/sqrt(64)
  const int q_hi = q0 + 15;

  for (int kt = 0; kt <= q_hi; kt += 32) {
    // K B-fragments: column n = key row (contiguous along d in [B,H,T,D])
    v16bf kf[2][2];
#pragma unroll
    for (int knt = 0; knt < 2; ++knt)
#pragma unroll
      for (int c = 0; c < 2; ++c) {
        const bf16_t* s = kp + (size_t)(kt + knt * 16 + ll) * DHEAD + c * 32 + lh * 8;
        kf[knt][c] = cat16(*(const v8bf*)s, *(const v8bf*)(s + 16));
      }

    // S = Q K^T for 16x32 key slab (two 16x16 tiles, K=64 over d)
    v8f sacc[2];
#pragma unroll
    for (int knt = 0; knt < 2; ++knt) {
      sacc[knt] = v8f_zero();
      sacc[knt] = __builtin_amdgcn_wmma_f32_16x16x32_bf16(
          false, qf[0], false, kf[knt][0], (short)0, sacc[knt], false, false);
      sacc[knt] = __builtin_amdgcn_wmma_f32_16x16x32_bf16(
          false, qf[1], false, kf[knt][1], (short)0, sacc[knt], false, false);
    }

    // scale + causal mask
    float p[2][8];
#pragma unroll
    for (int knt = 0; knt < 2; ++knt)
#pragma unroll
      for (int r = 0; r < 8; ++r) {
        const int qi = q0 + r + 8 * lh;
        const int ki = kt + knt * 16 + ll;
        const float sv = sacc[knt][r] * scale;
        p[knt][r] = (ki <= qi) ? sv : -1e30f;
      }

    // online softmax: per-row reduction across 16 lanes of each half-wave
#pragma unroll
    for (int r = 0; r < 8; ++r) {
      float rm = fmaxf(p[0][r], p[1][r]);
#pragma unroll
      for (int msk = 1; msk < 16; msk <<= 1)
        rm = fmaxf(rm, __shfl_xor(rm, msk, 32));
      const float mnew = fmaxf(mi[r], rm);
      const float corr = __expf(mi[r] - mnew);
      const float e0 = __expf(p[0][r] - mnew);
      const float e1 = __expf(p[1][r] - mnew);
      p[0][r] = e0;
      p[1][r] = e1;
      float rs = e0 + e1;
#pragma unroll
      for (int msk = 1; msk < 16; msk <<= 1)
        rs += __shfl_xor(rs, msk, 32);
      li[r] = li[r] * corr + rs;
      mi[r] = mnew;
#pragma unroll
      for (int dt = 0; dt < 4; ++dt) o[dt][r] *= corr;
    }

    // redistribute P (C layout) -> A-fragment layout through per-wave LDS tile
#pragma unroll
    for (int knt = 0; knt < 2; ++knt)
#pragma unroll
      for (int r = 0; r < 8; ++r)
        Ps[wid][r + 8 * lh][knt * 16 + ll] = (__bf16)p[knt][r];
    asm volatile("s_wait_dscnt 0x0" ::: "memory");
    v16bf pf;
    {
      const bf16_t* s = &Ps[wid][ll][lh * 8];
      pf = cat16(*(const v8bf*)s, *(const v8bf*)(s + 16));
    }

    // O += P V : V^T layout [B,H,D,T] makes B-frag columns contiguous
#pragma unroll
    for (int dt = 0; dt < 4; ++dt) {
      const bf16_t* s = vp + (size_t)(dt * 16 + ll) * TSEQ + kt + lh * 8;
      v16bf vf = cat16(*(const v8bf*)s, *(const v8bf*)(s + 16));
      o[dt] = __builtin_amdgcn_wmma_f32_16x16x32_bf16(
          false, pf, false, vf, (short)0, o[dt], false, false);
    }
  }

  // write y [B,T,C] bf16
#pragma unroll
  for (int dt = 0; dt < 4; ++dt)
#pragma unroll
    for (int r = 0; r < 8; ++r) {
      const int t = q0 + r + 8 * lh;
      const float val = o[dt][r] / li[r];
      yb[((size_t)(b * TSEQ + t)) * CDIM + h * DHEAD + dt * 16 + ll] = (__bf16)val;
    }
}

// ---------------------------------------------------------------------------
// Kernel 3: out = yb @ wptT  (bf16 A and B via async-LDS double buffering)
// ---------------------------------------------------------------------------
__global__ __launch_bounds__(256) void proj_gemm_kernel(
    const bf16_t* __restrict__ yb, const bf16_t* __restrict__ wpt,
    float* __restrict__ out)
{
  constexpr int K = CDIM;
  constexpr int N = CDIM;
  __shared__ __align__(16) bf16_t As[2][128][32];
  __shared__ __align__(16) bf16_t Bs[2][64][32];

  const int tid  = threadIdx.x;
  const int lane = tid & 31;
  const int wid  = tid >> 5;
  const int ll   = lane & 15;
  const int lh   = lane >> 4;
  const int waveM = wid & 3;
  const int waveN = wid >> 2;
  const int m0 = blockIdx.y * 128;
  const int n0 = blockIdx.x * 64;

  v8f acc[2][2];
#pragma unroll
  for (int i = 0; i < 2; ++i)
#pragma unroll
    for (int j = 0; j < 2; ++j) acc[i][j] = v8f_zero();

  auto stage = [&](int k0, int buf) {
#pragma unroll
    for (int p = 0; p < 2; ++p) {
      const int row = p * 64 + (tid >> 2);
      const int kc  = (tid & 3) * 8;
      async_copy16(yb + (size_t)(m0 + row) * K + k0 + kc, &As[buf][row][kc]);
    }
    const int n  = tid >> 2;
    const int kc = (tid & 3) * 8;
    async_copy16(wpt + (size_t)(n0 + n) * K + k0 + kc, &Bs[buf][n][kc]);
  };

  stage(0, 0);
  for (int k0 = 0; k0 < K; k0 += 32) {
    const int cur = (k0 >> 5) & 1;
    if (k0 + 32 < K) { stage(k0 + 32, cur ^ 1); wait_async3(); }
    else             { wait_async0(); }
    __syncthreads();

    v16bf af[2], bfr[2];
#pragma unroll
    for (int mt = 0; mt < 2; ++mt) {
      const bf16_t* s = &As[cur][waveM * 32 + mt * 16 + ll][lh * 8];
      af[mt] = cat16(*(const v8bf*)s, *(const v8bf*)(s + 16));
    }
#pragma unroll
    for (int nt = 0; nt < 2; ++nt) {
      const bf16_t* s = &Bs[cur][waveN * 32 + nt * 16 + ll][lh * 8];
      bfr[nt] = cat16(*(const v8bf*)s, *(const v8bf*)(s + 16));
    }
#pragma unroll
    for (int mt = 0; mt < 2; ++mt)
#pragma unroll
      for (int nt = 0; nt < 2; ++nt)
        acc[mt][nt] = __builtin_amdgcn_wmma_f32_16x16x32_bf16(
            false, af[mt], false, bfr[nt], (short)0, acc[mt][nt], false, false);
    __syncthreads();
  }

#pragma unroll
  for (int mt = 0; mt < 2; ++mt)
#pragma unroll
    for (int nt = 0; nt < 2; ++nt) {
      const int n = n0 + waveN * 32 + nt * 16 + ll;
#pragma unroll
      for (int r = 0; r < 8; ++r) {
        const int m = m0 + waveM * 32 + mt * 16 + r + 8 * lh;
        out[(size_t)m * N + n] = acc[mt][nt][r];
      }
    }
}

// ---------------------------------------------------------------------------
extern "C" void kernel_launch(void* const* d_in, const int* in_sizes, int n_in,
                              void* d_out, int out_size, void* d_ws, size_t ws_size,
                              hipStream_t stream) {
  (void)in_sizes; (void)n_in; (void)out_size; (void)ws_size;
  const float* x      = (const float*)d_in[0];
  const float* w_attn = (const float*)d_in[1];
  const float* w_proj = (const float*)d_in[2];
  float* out = (float*)d_out;

  const size_t xElems  = (size_t)BATCH * TSEQ * CDIM;       // 4Mi
  const size_t waElems = (size_t)CDIM * 3 * CDIM;           // 3Mi
  const size_t wpElems = (size_t)CDIM * CDIM;               // 1Mi
  const size_t per     = (size_t)BATCH * NHEAD * TSEQ * DHEAD; // 4Mi

  bf16_t* xb  = (bf16_t*)d_ws;
  bf16_t* wat = xb + xElems;
  bf16_t* wpt = wat + waElems;
  bf16_t* qb  = wpt + wpElems;
  bf16_t* kb  = qb + per;
  bf16_t* vt  = kb + per;
  bf16_t* yb  = vt + per;   // total 48 MiB of bf16 workspace

  cvt_kernel<<<(int)((xElems / 8 + 255) / 256), 256, 0, stream>>>(x, xb, (int)(xElems / 8));
  wtrans_kernel<<<dim3(3 * CDIM / 32, CDIM / 32), 256, 0, stream>>>(w_attn, wat, CDIM, 3 * CDIM);
  wtrans_kernel<<<dim3(CDIM / 32, CDIM / 32), 256, 0, stream>>>(w_proj, wpt, CDIM, CDIM);
  qkv_gemm_kernel<<<dim3(3 * CDIM / 64, (BATCH * TSEQ) / 128), 256, 0, stream>>>(
      xb, wat, qb, kb, vt);
  flash_attn_kernel<<<dim3(TSEQ / 64, BATCH * NHEAD), 128, 0, stream>>>(
      qb, kb, vt, yb);
  proj_gemm_kernel<<<dim3(CDIM / 64, (BATCH * TSEQ) / 128), 256, 0, stream>>>(
      yb, wpt, out);
}
